// RecurrentGCN_73555609911748
// MI455X (gfx1250) — compile-verified, matrix operand order checked by reading, not run on hardware
//
#include <hip/hip_runtime.h>
#include <hip/hip_bf16.h>
#include <math.h>

typedef __attribute__((ext_vector_type(16))) _Float16 v16h;
typedef __attribute__((ext_vector_type(8)))  _Float16 v8h;
typedef __attribute__((ext_vector_type(8)))  float    v8f;

#define BN_EPS 1e-5f

__device__ __forceinline__ float sigmoidf_(float x) { return 1.0f / (1.0f + expf(-x)); }

// K index base for 16-bit WMMA A/B fragments (ISA 7.12.2):
// VGPR j in 0..3 -> K = g*8 + 2j ; j in 4..7 -> K = 16 + g*8 + 2(j-4)
__device__ __forceinline__ int kbase_of(int j, int g) {
    return (j < 4) ? (g * 8 + 2 * j) : (16 + g * 8 + 2 * (j - 4));
}

// A fragment: 16x32 f16 tile. Lane's halves are two contiguous 8-half runs ->
// two b128 loads.
__device__ __forceinline__ v16h load_a_frag(const _Float16* __restrict__ base, int ldk,
                                            int row, int kofs, int g) {
    const _Float16* p = base + (size_t)row * ldk + kofs + g * 8;
    v8h lo = *(const v8h*)(p);
    v8h hi = *(const v8h*)(p + 16);
    v16h a;
#pragma unroll
    for (int i = 0; i < 8; ++i) { a[i] = lo[i]; a[8 + i] = hi[i]; }
    return a;
}

// ---------------- weight pre-pack kernels (run once per launch; tiny) ----------------
// Pack f32 weights into per-lane f16 B-fragment layout: 16 halves per (tile, lane).

// LSTM weights stored [n_out, Ktot] row-major; gate tiles {0,1,4,5,6,7} (f-gate dead).
__global__ void k_pack_lstm(const float* __restrict__ W, int Ktot, int kchunks,
                            _Float16* __restrict__ outp) {
    int tid = blockIdx.x * blockDim.x + threadIdx.x;
    if (tid >= 6 * kchunks * 32) return;
    int lane = tid & 31;
    int rest = tid >> 5;
    int kc = rest % kchunks;
    int q = rest / kchunks;
    int nt = (q < 2) ? q : q + 2;            // 0,1,4,5,6,7
    int g = lane >> 4, l = lane & 15;
    int ncol = nt * 16 + l;
    _Float16* o = outp + (size_t)((q * kchunks + kc) * 32 + lane) * 16;
    const float* p = W + (size_t)ncol * Ktot + kc * 32;
#pragma unroll
    for (int j = 0; j < 8; ++j) {
        int kb = kbase_of(j, g);
        o[2 * j] = (_Float16)p[kb];
        o[2 * j + 1] = (_Float16)p[kb + 1];
    }
}

// GCN2 weight stored [K, n_out] row-major (K strided in memory).
__global__ void k_pack_gcn2(const float* __restrict__ W2, _Float16* __restrict__ outp) {
    int tid = blockIdx.x * blockDim.x + threadIdx.x;
    if (tid >= 2 * 32) return;
    int lane = tid & 31;
    int nt = tid >> 5;
    int g = lane >> 4, l = lane & 15;
    int ncol = nt * 16 + l;
    _Float16* o = outp + (size_t)tid * 16;
#pragma unroll
    for (int j = 0; j < 8; ++j) {
        int kb = kbase_of(j, g);
        o[2 * j] = (_Float16)W2[(size_t)kb * 32 + ncol];
        o[2 * j + 1] = (_Float16)W2[(size_t)(kb + 1) * 32 + ncol];
    }
}

// ---------------- degree / normalization ----------------

__global__ void k_deg_init(float* __restrict__ deg, int n) {
    int i = blockIdx.x * blockDim.x + threadIdx.x;
    if (i < n) deg[i] = 1.0f;  // self-loop weight
}

__global__ void k_deg_edges(const int* __restrict__ dst, const float* __restrict__ ew,
                            float* __restrict__ deg, int e) {
    int t = blockIdx.x * blockDim.x + threadIdx.x;
    if (t < e) atomicAdd(&deg[dst[t]], ew[t]);
}

__global__ void k_dinv(float* __restrict__ deg, int n) {
    int i = blockIdx.x * blockDim.x + threadIdx.x;
    if (i < n) {
        float d = deg[i];
        deg[i] = (d > 0.0f) ? rsqrtf(d) : 0.0f;
    }
}

// ---------------- GCN layer 1: xw = x @ W1 ; acc = dinv^2 * xw (self loop) ----------------

__global__ void k_xw1(const float* __restrict__ x, const float* __restrict__ W1,
                      const float* __restrict__ dinv, float* __restrict__ xw,
                      float* __restrict__ acc, int n) {
    int idx = blockIdx.x * blockDim.x + threadIdx.x;
    if (idx >= n * 32) return;
    int j = idx & 31, i = idx >> 5;
    const float* xi = x + (size_t)i * 4;
    float s = xi[0] * W1[j] + xi[1] * W1[32 + j] + xi[2] * W1[64 + j] + xi[3] * W1[96 + j];
    xw[idx] = s;
    float di = dinv[i];
    acc[idx] = di * di * s;
}

// ---------------- edge scatter: acc[d] += dinv[s]*w*dinv[d] * xw[s] ----------------
// 4 lanes per edge: quad of lanes covers one 128B xw row contiguously.

__global__ void k_scatter(const int* __restrict__ src, const int* __restrict__ dst,
                          const float* __restrict__ ew, const float* __restrict__ dinv,
                          const float* __restrict__ xw, float* __restrict__ acc, int e) {
    int t = blockIdx.x * blockDim.x + threadIdx.x;
    int edge = t >> 2;
    if (edge >= e) return;
    int q = t & 3;
    int s = src[edge], d = dst[edge];
    float nrm = dinv[s] * ew[edge] * dinv[d];
    const float4* xs = (const float4*)(xw + (size_t)s * 32 + q * 8);
    float* od = acc + (size_t)d * 32 + q * 8;
    float4 v0 = xs[0], v1 = xs[1];
    atomicAdd(od + 0, nrm * v0.x);
    atomicAdd(od + 1, nrm * v0.y);
    atomicAdd(od + 2, nrm * v0.z);
    atomicAdd(od + 3, nrm * v0.w);
    atomicAdd(od + 4, nrm * v1.x);
    atomicAdd(od + 5, nrm * v1.y);
    atomicAdd(od + 6, nrm * v1.z);
    atomicAdd(od + 7, nrm * v1.w);
}

// ---------------- bias + relu + BN -> f16 outputs for WMMA stages ----------------

__global__ void k_post(const float* __restrict__ acc, const float* __restrict__ b,
                       const float* __restrict__ gamma, const float* __restrict__ beta,
                       const float* __restrict__ mean, const float* __restrict__ var,
                       _Float16* __restrict__ hf16, _Float16* __restrict__ skip,
                       int skip_off, int n) {
    int idx = blockIdx.x * blockDim.x + threadIdx.x;
    if (idx >= n * 32) return;
    int j = idx & 31, i = idx >> 5;
    float v = fmaxf(acc[idx] + b[j], 0.0f);
    float bn = (v - mean[j]) * rsqrtf(var[j] + BN_EPS) * gamma[j] + beta[j];
    _Float16 h = (_Float16)bn;
    if (hf16) hf16[idx] = h;
    skip[(size_t)i * 64 + skip_off + j] = h;
}

// ---------------- GCN layer 2 GEMM via WMMA: xw2 = h1 @ W2 ; acc = dinv^2 * xw2 ----------------

__global__ void k_gcn2_wmma(const _Float16* __restrict__ h1f16,
                            const _Float16* __restrict__ bpack,
                            const float* __restrict__ dinv, float* __restrict__ xw,
                            float* __restrict__ acc, int n) {
    int wave = (blockIdx.x * blockDim.x + threadIdx.x) >> 5;
    int row0 = wave * 16;
    if (row0 >= n) return;  // wave-uniform: EXEC all ones for WMMA
    int lane = threadIdx.x & 31;
    int g = lane >> 4, l = lane & 15;
    v16h a = load_a_frag(h1f16, 32, row0 + l, 0, g);
#pragma unroll
    for (int nt = 0; nt < 2; ++nt) {
        v16h b = *(const v16h*)(bpack + (size_t)(nt * 32 + lane) * 16);
        v8f c = {};
        c = __builtin_amdgcn_wmma_f32_16x16x32_f16(false, a, false, b, (short)0, c, false, false);
        int col = nt * 16 + l;
#pragma unroll
        for (int r = 0; r < 8; ++r) {
            int rowi = row0 + r + 8 * g;
            float v = c[r];
            xw[(size_t)rowi * 32 + col] = v;
            float di = dinv[rowi];
            acc[(size_t)rowi * 32 + col] = di * di * v;
        }
    }
}

// ---------------- LSTM step 1 via WMMA: gates = skip[N,64] @ wih^T, fused i/g/o ----------------
// f-gate skipped: with (h0,c0)=0, sigmoid(f)*c0 == 0 never reaches the output.

__global__ void k_lstm1_wmma(const _Float16* __restrict__ skip,
                             const _Float16* __restrict__ bpack,
                             const float* __restrict__ bih, const float* __restrict__ bhh,
                             float* __restrict__ H1, _Float16* __restrict__ H1h, int n) {
    int wave = (blockIdx.x * blockDim.x + threadIdx.x) >> 5;
    int row0 = wave * 16;
    if (row0 >= n) return;
    int lane = threadIdx.x & 31;
    int g = lane >> 4, l = lane & 15;
    v16h a0 = load_a_frag(skip, 64, row0 + l, 0, g);
    v16h a1 = load_a_frag(skip, 64, row0 + l, 32, g);
    v8f c[6];
    float bi[6];
#pragma unroll
    for (int q = 0; q < 6; ++q) {
        int nt = (q < 2) ? q : q + 2;
        int nc = nt * 16 + l;
        v16h b0 = *(const v16h*)(bpack + (size_t)((q * 2 + 0) * 32 + lane) * 16);
        v8f cc = {};
        cc = __builtin_amdgcn_wmma_f32_16x16x32_f16(false, a0, false, b0, (short)0, cc, false, false);
        v16h b1 = *(const v16h*)(bpack + (size_t)((q * 2 + 1) * 32 + lane) * 16);
        cc = __builtin_amdgcn_wmma_f32_16x16x32_f16(false, a1, false, b1, (short)0, cc, false, false);
        c[q] = cc;
        bi[q] = bih[nc] + bhh[nc];
    }
#pragma unroll
    for (int t = 0; t < 2; ++t) {
        int qi = t, qg = 2 + t, qo = 4 + t;
#pragma unroll
        for (int r = 0; r < 8; ++r) {
            float iv = c[qi][r] + bi[qi];
            float gv = c[qg][r] + bi[qg];
            float ov = c[qo][r] + bi[qo];
            float cv = sigmoidf_(iv) * tanhf(gv);
            float hv = sigmoidf_(ov) * tanhf(cv);
            size_t idx = (size_t)(row0 + r + 8 * g) * 32 + t * 16 + l;
            H1[idx] = hv;
            H1h[idx] = (_Float16)hv;
        }
    }
}

// ---------------- LSTM step 2 via WMMA: gates = H1[N,32] @ wih^T ----------------

__global__ void k_lstm2_wmma(const _Float16* __restrict__ H1h,
                             const _Float16* __restrict__ bpack,
                             const float* __restrict__ bih, const float* __restrict__ bhh,
                             float* __restrict__ H2, int n) {
    int wave = (blockIdx.x * blockDim.x + threadIdx.x) >> 5;
    int row0 = wave * 16;
    if (row0 >= n) return;
    int lane = threadIdx.x & 31;
    int g = lane >> 4, l = lane & 15;
    v16h a = load_a_frag(H1h, 32, row0 + l, 0, g);
    v8f c[6];
    float bi[6];
#pragma unroll
    for (int q = 0; q < 6; ++q) {
        int nt = (q < 2) ? q : q + 2;
        int nc = nt * 16 + l;
        v16h b = *(const v16h*)(bpack + (size_t)(q * 32 + lane) * 16);
        v8f cc = {};
        cc = __builtin_amdgcn_wmma_f32_16x16x32_f16(false, a, false, b, (short)0, cc, false, false);
        c[q] = cc;
        bi[q] = bih[nc] + bhh[nc];
    }
#pragma unroll
    for (int t = 0; t < 2; ++t) {
        int qi = t, qg = 2 + t, qo = 4 + t;
#pragma unroll
        for (int r = 0; r < 8; ++r) {
            float iv = c[qi][r] + bi[qi];
            float gv = c[qg][r] + bi[qg];
            float ov = c[qo][r] + bi[qo];
            float cv = sigmoidf_(iv) * tanhf(gv);
            float hv = sigmoidf_(ov) * tanhf(cv);
            H2[(size_t)(row0 + r + 8 * g) * 32 + t * 16 + l] = hv;
        }
    }
}

// ---------------- head: out = relu(concat(H1,H2,x)) @ lin_w^T + lin_b ----------------

__global__ void k_head(const float* __restrict__ H1, const float* __restrict__ H2,
                       const float* __restrict__ x, const float* __restrict__ lw,
                       const float* __restrict__ lb, float* __restrict__ out, int n) {
    int i = blockIdx.x * blockDim.x + threadIdx.x;
    if (i >= n) return;
    float s = lb[0];
    const float4* p1 = (const float4*)(H1 + (size_t)i * 32);
    const float4* p2 = (const float4*)(H2 + (size_t)i * 32);
#pragma unroll
    for (int q = 0; q < 8; ++q) {
        float4 v = p1[q];
        s += fmaxf(v.x, 0.f) * lw[4 * q + 0] + fmaxf(v.y, 0.f) * lw[4 * q + 1] +
             fmaxf(v.z, 0.f) * lw[4 * q + 2] + fmaxf(v.w, 0.f) * lw[4 * q + 3];
    }
#pragma unroll
    for (int q = 0; q < 8; ++q) {
        float4 v = p2[q];
        s += fmaxf(v.x, 0.f) * lw[32 + 4 * q + 0] + fmaxf(v.y, 0.f) * lw[32 + 4 * q + 1] +
             fmaxf(v.z, 0.f) * lw[32 + 4 * q + 2] + fmaxf(v.w, 0.f) * lw[32 + 4 * q + 3];
    }
    const float* xi = x + (size_t)i * 4;
#pragma unroll
    for (int t = 0; t < 4; ++t) s += fmaxf(xi[t], 0.f) * lw[64 + t];
    out[i] = s;
}

extern "C" void kernel_launch(void* const* d_in, const int* in_sizes, int n_in,
                              void* d_out, int out_size, void* d_ws, size_t ws_size,
                              hipStream_t stream) {
    const float* x       = (const float*)d_in[0];
    const int*   ei      = (const int*)d_in[1];
    const float* ew      = (const float*)d_in[2];
    const float* gcn1_w  = (const float*)d_in[3];
    const float* gcn1_b  = (const float*)d_in[4];
    const float* gcn2_w  = (const float*)d_in[5];
    const float* gcn2_b  = (const float*)d_in[6];
    const float* bn1_g   = (const float*)d_in[7];
    const float* bn1_be  = (const float*)d_in[8];
    const float* bn1_m   = (const float*)d_in[9];
    const float* bn1_v   = (const float*)d_in[10];
    const float* bn2_g   = (const float*)d_in[11];
    const float* bn2_be  = (const float*)d_in[12];
    const float* bn2_m   = (const float*)d_in[13];
    const float* bn2_v   = (const float*)d_in[14];
    const float* l1_wih  = (const float*)d_in[15];
    // d_in[16] lstm1_whh unused (zero initial hidden state)
    const float* l1_bih  = (const float*)d_in[17];
    const float* l1_bhh  = (const float*)d_in[18];
    const float* l2_wih  = (const float*)d_in[19];
    // d_in[20] lstm2_whh unused
    const float* l2_bih  = (const float*)d_in[21];
    const float* l2_bhh  = (const float*)d_in[22];
    const float* lin_w   = (const float*)d_in[23];
    const float* lin_b   = (const float*)d_in[24];

    const int n = in_sizes[0] / 4;   // 200000
    const int e = in_sizes[2];       // 3200000
    const int* src = ei;
    const int* dst = ei + e;

    // workspace carve (floats; every block a multiple of 8 floats -> 32B aligned)
    float* wsf = (float*)d_ws;
    size_t o = 0;
    float* dinv = wsf + o; o += (size_t)n;        // deg -> dinv in place
    float* xw   = wsf + o; o += (size_t)n * 32;
    float* acc  = wsf + o; o += (size_t)n * 32;
    float* H1   = wsf + o; o += (size_t)n * 32;
    float* H2   = wsf + o; o += (size_t)n * 32;
    _Float16* skiph = (_Float16*)(wsf + o); o += (size_t)n * 32;  // n*64 halves
    _Float16* h1h   = (_Float16*)(wsf + o); o += (size_t)n * 16;  // n*32 halves
    _Float16* H1h   = (_Float16*)(wsf + o); o += (size_t)n * 16;  // n*32 halves
    _Float16* bp1   = (_Float16*)(wsf + o); o += 3072;            // lstm1: 6*2*32*16 halves
    _Float16* bp2   = (_Float16*)(wsf + o); o += 1536;            // lstm2: 6*1*32*16 halves
    _Float16* bpg   = (_Float16*)(wsf + o); o += 512;             // gcn2:  2*32*16 halves

    const int B = 256;
    int gN   = (n + B - 1) / B;
    int gNF  = (n * 32 + B - 1) / B;
    int gE4  = (e * 4 + B - 1) / B;
    int gE   = (e + B - 1) / B;
    int tiles = (n + 15) / 16;
    int gW   = (tiles * 32 + B - 1) / B;
    float* out = (float*)d_out;

    // weight pre-pack (tiny, L2-resident afterwards)
    k_pack_lstm<<<2, B, 0, stream>>>(l1_wih, 64, 2, bp1);
    k_pack_lstm<<<1, B, 0, stream>>>(l2_wih, 32, 1, bp2);
    k_pack_gcn2<<<1, B, 0, stream>>>(gcn2_w, bpg);

    // degree + symmetric normalization
    k_deg_init<<<gN, B, 0, stream>>>(dinv, n);
    k_deg_edges<<<gE, B, 0, stream>>>(dst, ew, dinv, e);
    k_dinv<<<gN, B, 0, stream>>>(dinv, n);

    // GCN layer 1
    k_xw1<<<gNF, B, 0, stream>>>(x, gcn1_w, dinv, xw, acc, n);
    k_scatter<<<gE4, B, 0, stream>>>(src, dst, ew, dinv, xw, acc, e);
    k_post<<<gNF, B, 0, stream>>>(acc, gcn1_b, bn1_g, bn1_be, bn1_m, bn1_v,
                                  h1h, skiph, 0, n);

    // GCN layer 2 (WMMA GEMM)
    k_gcn2_wmma<<<gW, B, 0, stream>>>(h1h, bpg, dinv, xw, acc, n);
    k_scatter<<<gE4, B, 0, stream>>>(src, dst, ew, dinv, xw, acc, e);
    k_post<<<gNF, B, 0, stream>>>(acc, gcn2_b, bn2_g, bn2_be, bn2_m, bn2_v,
                                  (_Float16*)nullptr, skiph, 32, n);

    // LSTM steps (WMMA GEMMs, f-gate elided)
    k_lstm1_wmma<<<gW, B, 0, stream>>>(skiph, bp1, l1_bih, l1_bhh, H1, H1h, n);
    k_lstm2_wmma<<<gW, B, 0, stream>>>(H1h, bp2, l2_bih, l2_bhh, H2, n);

    // head
    k_head<<<gN, B, 0, stream>>>(H1, H2, x, lin_w, lin_b, out, n);
}
